// EncoderLayer_21406117003700
// MI455X (gfx1250) — matrix operand order repaired and checked by reading, not compiled, not run on hardware
//
#include <hip/hip_runtime.h>

typedef __attribute__((ext_vector_type(16))) _Float16 v16h;
typedef __attribute__((ext_vector_type(8)))  _Float16 v8h;
typedef __attribute__((ext_vector_type(8)))  float    v8f;

#define IN_DIM 128
#define HC     128
#define HEADS  4
#define CH     32

// ---------- order-preserving float<->uint encoding for atomicMax ----------
__device__ __forceinline__ unsigned f32_ord(float f) {
  unsigned u = __float_as_uint(f);
  return (u & 0x80000000u) ? ~u : (u | 0x80000000u);
}
__device__ __forceinline__ float ord_f32(unsigned u) {
  return __uint_as_float((u & 0x80000000u) ? (u & 0x7FFFFFFFu) : ~u);
}

// ---------------------- generic zero fill ----------------------
__global__ void k_zero(float* __restrict__ p, long long n) {
  long long t = (long long)blockIdx.x * blockDim.x + threadIdx.x;
  if (t < n) p[t] = 0.0f;
}

// ---------------------- f32 -> f16 streaming convert (n % 8 == 0) ----------------------
__global__ void k_cvt_f16(const float* __restrict__ src, _Float16* __restrict__ dst,
                          long long n) {
  long long t = ((long long)blockIdx.x * blockDim.x + threadIdx.x) * 8;
  if (t + 8 > n) return;
  float4 f0 = *(const float4*)(src + t);
  float4 f1 = *(const float4*)(src + t + 4);
  v8h h;
  h[0] = (_Float16)f0.x; h[1] = (_Float16)f0.y;
  h[2] = (_Float16)f0.z; h[3] = (_Float16)f0.w;
  h[4] = (_Float16)f1.x; h[5] = (_Float16)f1.y;
  h[6] = (_Float16)f1.z; h[7] = (_Float16)f1.w;
  *(v8h*)(dst + t) = h;
}

// ---------------------- coef[h] = sum_c W_edge[h*C+c]*att_edge[h*C+c] ----------------------
__global__ void k_coef(const float* __restrict__ W_edge,
                       const float* __restrict__ att_edge,
                       float* __restrict__ coef) {
  int h = threadIdx.x;
  if (h < HEADS) {
    float s = 0.0f;
    #pragma unroll
    for (int c = 0; c < CH; ++c) s += W_edge[h * CH + c] * att_edge[h * CH + c];
    coef[h] = s;
  }
}

// ---------------------- WMMA GEMM: xp = x @ W_src^T (f16 inputs, f32 accum) --------------
// Block = 256 threads = 8 waves; each wave owns one 16-wide column tile of a
// 16x128 output tile. A tile (16x128 f16 = 4 KB) is staged to LDS with one
// global_load_async_to_lds_b128 per thread, then read back via ds_load_b128.
__global__ __launch_bounds__(256) void k_gemm_xp(const _Float16* __restrict__ xh,
                                                 const _Float16* __restrict__ wh,
                                                 float* __restrict__ xp, int N) {
  __shared__ _Float16 As[16 * IN_DIM];          // 4 KB A tile

  const int t       = threadIdx.x;
  const int lane    = t & 31;
  const int wave    = t >> 5;                   // 0..7 -> column tile
  const int row0    = blockIdx.x * 16;
  const int colBase = wave * 16;
  const int n       = lane & 15;                // col within tile / A row
  const int hi      = lane >> 4;                // lane half (0/1)

  // ---- async stage A tile: thread t moves halfs [t*8, t*8+8) (16 bytes) ----
  {
    int elt = t * 8;
    int r   = elt >> 7;                         // row within tile (t/16)
    int k   = elt & 127;                        // K offset ((t%16)*8)
    int gr  = row0 + r; if (gr >= N) gr = N - 1;
    const _Float16* gp = xh + (long long)gr * IN_DIM + k;
    unsigned ldsoff = (unsigned)(uintptr_t)&As[elt];
    asm volatile("global_load_async_to_lds_b128 %0, %1, off"
                 :: "v"(ldsoff), "v"((unsigned long long)(uintptr_t)gp)
                 : "memory");
  }
  asm volatile("s_wait_asynccnt 0x0" ::: "memory");
  __syncthreads();

  const int col = colBase + n;
  v8f acc = {};
  #pragma unroll
  for (int k0 = 0; k0 < IN_DIM; k0 += 32) {
    // A fragment 16x32: lane<16 -> K = k0+{0..15}; lane>=16 -> K = k0+{16..31}
    const _Float16* ap = As + n * IN_DIM + k0 + hi * 16;
    v8h al = *(const v8h*)(ap);                 // ds_load_b128
    v8h ah = *(const v8h*)(ap + 8);             // ds_load_b128
    v16h a = __builtin_shufflevector(al, ah, 0, 1, 2, 3, 4, 5, 6, 7,
                                             8, 9, 10, 11, 12, 13, 14, 15);
    // B fragment 32x16: B[k][n] = W[col][k]; lanes 0-15: K=k0..k0+15,
    // lanes 16-31: K=k0+16..k0+31 -> 16 contiguous halfs
    v16h b = *(const v16h*)(wh + (long long)col * IN_DIM + k0 + hi * 16);

    acc = __builtin_amdgcn_wmma_f32_16x16x32_f16(false, a, false, b,
                                                 (short)0, acc, false, false);
  }
  // D layout: VGPR v -> row (v + 8*hi), col n. Fast path: whole tile in-bounds
  // (single uniform branch; N % 16 == 0 makes this always true here).
  if (row0 + 16 <= N) {
    float* outp = xp + (long long)(row0 + 8 * hi) * HC + colBase + n;
    #pragma unroll
    for (int v = 0; v < 8; ++v) outp[(long long)v * HC] = acc[v];
  } else {
    #pragma unroll
    for (int v = 0; v < 8; ++v) {
      int r = row0 + v + 8 * hi;
      if (r < N) xp[(long long)r * HC + colBase + n] = acc[v];
    }
  }
}

// ---------------------- a_src/a_dst per (node, head) ----------------------
__global__ void k_attdot(const float* __restrict__ xp,
                         const float* __restrict__ att_src,
                         const float* __restrict__ att_dst,
                         float* __restrict__ a_src, float* __restrict__ a_dst, int N) {
  long long t = (long long)blockIdx.x * blockDim.x + threadIdx.x;  // t = i*H + h
  if (t >= (long long)N * HEADS) return;
  int i = (int)(t >> 2), h = (int)(t & 3);
  const float* row = xp + (long long)i * HC + h * CH;
  const float* as  = att_src + h * CH;
  const float* ad  = att_dst + h * CH;
  float s = 0.0f, d = 0.0f;
  #pragma unroll
  for (int c = 0; c < CH; ++c) { s += row[c] * as[c]; d += row[c] * ad[c]; }
  a_src[t] = s; a_dst[t] = d;
}

// ---------------------- degree / weight-sum per destination ----------------------
__global__ void k_deg(const int* __restrict__ ei, const float* __restrict__ ew,
                      float* __restrict__ deg, float* __restrict__ wsum, int E, int N) {
  int e = blockIdx.x * blockDim.x + threadIdx.x;
  if (e >= E) return;
  int d = ei[E + e];                    // edge_index[1][e]
  if ((unsigned)d >= (unsigned)N) return;
  atomicAdd(&deg[d], 1.0f);
  atomicAdd(&wsum[d], ew[e]);
}

// ---------------------- loop_w + m_enc init ----------------------
__global__ void k_loop(const float* __restrict__ deg, const float* __restrict__ wsum,
                       float* __restrict__ loop_w, unsigned* __restrict__ m_enc, int N) {
  int i = blockIdx.x * blockDim.x + threadIdx.x;
  if (i >= N) return;
  float dg = deg[i];
  loop_w[i] = (dg > 0.0f) ? (wsum[i] / fmaxf(dg, 1.0f)) : 0.0f;
  unsigned init = f32_ord(-3.0e38f);
  #pragma unroll
  for (int h = 0; h < HEADS; ++h) m_enc[i * HEADS + h] = init;
}

// ---------------------- alpha recompute helper ----------------------
__device__ __forceinline__ float edge_alpha(long long e, int E, int N,
    const int* __restrict__ ei, const float* __restrict__ ew,
    const float* __restrict__ loop_w,
    const float* __restrict__ a_src, const float* __restrict__ a_dst,
    const float* __restrict__ coef, int h, int* os, int* od) {
  int s, d; float w;
  if (e < E) {
    s = ei[e]; d = ei[E + e]; w = ew[e];
    if ((unsigned)s >= (unsigned)N) s = 0;
    if ((unsigned)d >= (unsigned)N) d = 0;
  } else {
    s = d = (int)(e - E); w = loop_w[s];
  }
  *os = s; *od = d;
  float al = a_src[s * HEADS + h] + a_dst[d * HEADS + h] + coef[h] * w;
  return (al > 0.0f) ? al : 0.2f * al;   // leaky_relu(0.2)
}

// ---------------------- pass 1: segment max of alpha ----------------------
__global__ void k_max(const int* __restrict__ ei, const float* __restrict__ ew,
                      const float* __restrict__ loop_w,
                      const float* __restrict__ a_src, const float* __restrict__ a_dst,
                      const float* __restrict__ coef,
                      unsigned* __restrict__ m_enc, int E, int N) {
  long long t = (long long)blockIdx.x * blockDim.x + threadIdx.x;  // t = e*H + h
  long long tot = (long long)(E + N) * HEADS;
  if (t >= tot) return;
  long long e = t >> 2; int h = (int)(t & 3);
  int s, d;
  float al = edge_alpha(e, E, N, ei, ew, loop_w, a_src, a_dst, coef, h, &s, &d);
  atomicMax(&m_enc[d * HEADS + h], f32_ord(al));
}

// ---------------------- pass 2: segment sum of exp(alpha - m) ----------------------
__global__ void k_sum(const int* __restrict__ ei, const float* __restrict__ ew,
                      const float* __restrict__ loop_w,
                      const float* __restrict__ a_src, const float* __restrict__ a_dst,
                      const float* __restrict__ coef,
                      const unsigned* __restrict__ m_enc, float* __restrict__ denom,
                      int E, int N) {
  long long t = (long long)blockIdx.x * blockDim.x + threadIdx.x;
  long long tot = (long long)(E + N) * HEADS;
  if (t >= tot) return;
  long long e = t >> 2; int h = (int)(t & 3);
  int s, d;
  float al = edge_alpha(e, E, N, ei, ew, loop_w, a_src, a_dst, coef, h, &s, &d);
  float m = ord_f32(m_enc[d * HEADS + h]);
  atomicAdd(&denom[d * HEADS + h], __expf(al - m));
}

// ---------------------- pass 3: weighted scatter (one wave per edge) ----------------------
__global__ __launch_bounds__(256) void k_scatter(
    const int* __restrict__ ei, const float* __restrict__ ew,
    const float* __restrict__ loop_w,
    const float* __restrict__ a_src, const float* __restrict__ a_dst,
    const float* __restrict__ coef,
    const unsigned* __restrict__ m_enc, const float* __restrict__ denom,
    const float* __restrict__ xp, float* __restrict__ out, int E, int N) {
  long long wid = ((long long)blockIdx.x * blockDim.x + threadIdx.x) >> 5;  // edge id
  int lane = threadIdx.x & 31;
  if (wid >= (long long)(E + N)) return;
  int h = lane >> 3;                       // lane owns channels [lane*4, lane*4+4)
  int s, d;
  float al = edge_alpha(wid, E, N, ei, ew, loop_w, a_src, a_dst, coef, h, &s, &d);
  float m   = ord_f32(m_enc[d * HEADS + h]);
  float dn  = denom[d * HEADS + h];
  float att = __expf(al - m) / dn;

  const float* srcp = xp + (long long)s * HC + lane * 4;
  float*       dstp = out + (long long)d * HC + lane * 4;
  __builtin_prefetch(srcp, 0, 0);
  float4 v = *(const float4*)srcp;
  atomicAdd(dstp + 0, att * v.x);
  atomicAdd(dstp + 1, att * v.y);
  atomicAdd(dstp + 2, att * v.z);
  atomicAdd(dstp + 3, att * v.w);
}

// ---------------------- bias + final leaky_relu(0.01), in place ----------------------
__global__ void k_bias(float* __restrict__ out, const float* __restrict__ bias, int N) {
  long long t = (long long)blockIdx.x * blockDim.x + threadIdx.x;
  if (t >= (long long)N * HC) return;
  float v = out[t] + bias[t & (HC - 1)];
  out[t] = (v > 0.0f) ? v : 0.01f * v;
}

extern "C" void kernel_launch(void* const* d_in, const int* in_sizes, int n_in,
                              void* d_out, int out_size, void* d_ws, size_t ws_size,
                              hipStream_t stream) {
  const float* x        = (const float*)d_in[0];
  const int*   ei       = (const int*)d_in[1];   // edge_index flattened [2,E]
  const float* ew       = (const float*)d_in[2];
  const float* W_src    = (const float*)d_in[3];
  const float* att_src  = (const float*)d_in[4];
  const float* att_dst  = (const float*)d_in[5];
  const float* W_edge   = (const float*)d_in[6];
  const float* att_edge = (const float*)d_in[7];
  const float* bias     = (const float*)d_in[8];
  float* out = (float*)d_out;

  const int N = in_sizes[0] / IN_DIM;
  const int E = in_sizes[2];

  // ---- workspace layout (float units) ----
  float* ws = (float*)d_ws;
  long long o = 0;
  float*    xp     = ws + o; o += (long long)N * HC;
  float*    a_src  = ws + o; o += (long long)N * HEADS;
  float*    a_dst  = ws + o; o += (long long)N * HEADS;
  float*    coef   = ws + o; o += 16;
  float*    zbase  = ws + o;                    // zero-init region start
  float*    deg    = ws + o; o += N;
  float*    wsum   = ws + o; o += N;
  float*    denom  = ws + o; o += (long long)N * HEADS;
  long long zcnt   = (ws + o) - zbase;
  float*    loop_w = ws + o; o += N;
  unsigned* m_enc  = (unsigned*)(ws + o); o += (long long)N * HEADS;
  _Float16* xh     = (_Float16*)(ws + o); o += (long long)N * (IN_DIM / 2);
  _Float16* wh     = (_Float16*)(ws + o); o += (HC * IN_DIM) / 2;

  const int T = 256;
  const long long nx = (long long)N * IN_DIM;
  const long long nw = (long long)HC * IN_DIM;
  // 1) zero accumulators + output
  k_zero<<<(int)((zcnt + T - 1) / T), T, 0, stream>>>(zbase, zcnt);
  k_zero<<<(int)(((long long)N * HC + T - 1) / T), T, 0, stream>>>(out, (long long)N * HC);
  // 2) f32 -> f16 preconversion of x and W_src
  k_cvt_f16<<<(int)((nx / 8 + T - 1) / T), T, 0, stream>>>(x, xh, nx);
  k_cvt_f16<<<(int)((nw / 8 + T - 1) / T), T, 0, stream>>>(W_src, wh, nw);
  // 3) per-head edge coefficient
  k_coef<<<1, 32, 0, stream>>>(W_edge, att_edge, coef);
  // 4) WMMA GEMM xp = x @ W_src^T
  k_gemm_xp<<<(N + 15) / 16, 256, 0, stream>>>(xh, wh, xp, N);
  // 5) attention dot products
  k_attdot<<<(int)(((long long)N * HEADS + T - 1) / T), T, 0, stream>>>(
      xp, att_src, att_dst, a_src, a_dst, N);
  // 6) degree / weight sum
  k_deg<<<(E + T - 1) / T, T, 0, stream>>>(ei, ew, deg, wsum, E, N);
  // 7) self-loop weights + segment-max init
  k_loop<<<(N + T - 1) / T, T, 0, stream>>>(deg, wsum, loop_w, m_enc, N);
  // 8) segment max
  long long eh = (long long)(E + N) * HEADS;
  k_max<<<(int)((eh + T - 1) / T), T, 0, stream>>>(ei, ew, loop_w, a_src, a_dst, coef,
                                                   m_enc, E, N);
  // 9) segment sum of exponentials
  k_sum<<<(int)((eh + T - 1) / T), T, 0, stream>>>(ei, ew, loop_w, a_src, a_dst, coef,
                                                   m_enc, denom, E, N);
  // 10) weighted scatter: one wave per edge
  long long waves = (long long)(E + N);
  k_scatter<<<(int)((waves * 32 + T - 1) / T), T, 0, stream>>>(
      ei, ew, loop_w, a_src, a_dst, coef, m_enc, denom, xp, out, E, N);
  // 11) bias + output leaky_relu
  k_bias<<<(int)(((long long)N * HC + T - 1) / T), T, 0, stream>>>(out, bias, N);
}
